// Representation_300647710728
// MI455X (gfx1250) — compile-verified, matrix operand order checked by reading, not run on hardware
//
#include <hip/hip_runtime.h>
#include <hip/hip_bf16.h>

typedef __attribute__((ext_vector_type(16))) _Float16 v16h;
typedef __attribute__((ext_vector_type(8)))  float    v8f;

#define LVL 24
#define NN  4096
#define CST 10
#define DD  64
#define HH  128
#define SEQ (LVL * NN)          // 98304
#define FEATW 192               // 188 padded to 6 k-tiles of 32
#define KCAT  320               // 192 + 128
#define P1_SMEM (131072 + 65536 + 8*16*128*2 + 8*16*64*2)   // 245760 B
#define P2_SMEM (8*16*KCAT*2 + 8*16*128*4)                  // 147456 B

__device__ __forceinline__ float sigm(float x) { return 1.f / (1.f + __expf(-x)); }

union AF { v16h v; unsigned u[8]; };

// ---------------------------------------------------------------------------
// Weight packing into WMMA B-fragment layout:
// dst[((nt*kTiles + kt)*32 + lane)*16 + j] = W[n][k],
//   n = nt*16 + (lane&15),  k = kt*32 + j + 8*(lane>>4) + (j>=8 ? 8 : 0)
// ---------------------------------------------------------------------------
__global__ void k_pack_frag(const float* __restrict__ W, int K, int nTiles, int kTiles,
                            _Float16* __restrict__ dst) {
  int total = nTiles * kTiles * 32 * 16;
  for (int i = blockIdx.x * blockDim.x + threadIdx.x; i < total;
       i += gridDim.x * blockDim.x) {
    int j = i & 15, lane = (i >> 4) & 31, r = i >> 9;
    int kt = r % kTiles, nt = r / kTiles, hi = lane >> 4;
    int n = nt * 16 + (lane & 15);
    int k = kt * 32 + j + 8 * hi + ((j >= 8) ? 8 : 0);
    dst[i] = (_Float16)W[(size_t)n * K + k];
  }
}

// Packed concat weight for phase 2: K = [0..187]=Wih2, [188..191]=0, [192..319]=Whh2
__global__ void k_pack_w2(const float* __restrict__ Wih2, const float* __restrict__ Whh2,
                          _Float16* __restrict__ dst) {
  int total = 32 * 10 * 32 * 16;
  for (int i = blockIdx.x * blockDim.x + threadIdx.x; i < total;
       i += gridDim.x * blockDim.x) {
    int j = i & 15, lane = (i >> 4) & 31, r = i >> 9;
    int kt = r % 10, nt = r / 10, hi = lane >> 4;
    int n = nt * 16 + (lane & 15);
    int k = kt * 32 + j + 8 * hi + ((j >= 8) ? 8 : 0);
    float v = 0.f;
    if (k < 188)       v = Wih2[(size_t)n * 188 + k];
    else if (k >= 192) v = Whh2[(size_t)n * 128 + (k - 192)];
    dst[i] = (_Float16)v;
  }
}

__global__ void k_addvec(const float* __restrict__ a, const float* __restrict__ b,
                         float* __restrict__ o, int n) {
  int i = blockIdx.x * blockDim.x + threadIdx.x;
  if (i < n) o[i] = a[i] + b[i];
}

// ---------------------------------------------------------------------------
// Phase 1: fused 10-step LSTM over 98304 sequences. One wave owns 16 sequences.
// Weights live in LDS; h stays in per-wave LDS staging; c stays in registers.
// ---------------------------------------------------------------------------
__global__ void __launch_bounds__(256) k_lstm1(
    const float* __restrict__ cond,          // [SEQ][10][64]
    const _Float16* __restrict__ wih_pack,   // 32nt x 2kt frag layout (64 KB)
    const _Float16* __restrict__ whh_pack,   // 32nt x 4kt frag layout (128 KB)
    const float* __restrict__ bias1,         // 512
    float* __restrict__ h1out)               // [SEQ][128]
{
  extern __shared__ char smem[];
  _Float16* whh_lds = (_Float16*)smem;                       // 65536 halves
  _Float16* wih_lds = (_Float16*)(smem + 131072);            // 32768 halves
  int tid = threadIdx.x, wave = tid >> 5, lane = tid & 31;
  int hi = lane >> 4, lr = lane & 15;
  _Float16* h_lds = (_Float16*)(smem + 196608) + wave * (16 * 128);
  _Float16* x_lds = (_Float16*)(smem + 229376) + wave * (16 * 64);

  { // cooperative weight copy (global_load_b128 -> ds_store_b128)
    const uint4* s1 = (const uint4*)whh_pack; uint4* d1 = (uint4*)whh_lds;
    for (int i = tid; i < 131072 / 16; i += 256) d1[i] = s1[i];
    const uint4* s2 = (const uint4*)wih_pack; uint4* d2 = (uint4*)wih_lds;
    for (int i = tid; i < 65536 / 16; i += 256) d2[i] = s2[i];
  }
  for (int i = lane; i < 16 * 128; i += 32) h_lds[i] = (_Float16)0.f;
  __syncthreads();

  size_t seq0 = ((size_t)blockIdx.x * 8 + wave) * 16;

  float cf[8][8];
#pragma unroll
  for (int a = 0; a < 8; ++a)
#pragma unroll
    for (int e = 0; e < 8; ++e) cf[a][e] = 0.f;

#pragma unroll 1
  for (int t = 0; t < CST; ++t) {
    // stage x_t tile (f32 -> f16), per-wave region: intra-wave LDS ordering only
    for (int i = lane; i < 16 * 64; i += 32) {
      int r = i >> 6, c = i & 63;
      x_lds[i] = (_Float16)cond[((seq0 + r) * CST + t) * 64 + c];
    }
    // A-fragments for h (K=128: 4 k-tiles) and x (K=64: 2 k-tiles)
    AF ah[4], ax[2];
#pragma unroll
    for (int kt = 0; kt < 4; ++kt)
#pragma unroll
      for (int p = 0; p < 8; ++p) {
        int k = kt * 32 + 2 * p + 8 * hi + ((p >= 4) ? 8 : 0);
        ah[kt].u[p] = *(const unsigned*)&h_lds[lr * 128 + k];
      }
#pragma unroll
    for (int kt = 0; kt < 2; ++kt)
#pragma unroll
      for (int p = 0; p < 8; ++p) {
        int k = kt * 32 + 2 * p + 8 * hi + ((p >= 4) ? 8 : 0);
        ax[kt].u[p] = *(const unsigned*)&x_lds[lr * 64 + k];
      }

#pragma unroll
    for (int g8 = 0; g8 < 8; ++g8) {       // 8 column-groups of 16 hidden units
      v8f acc[4];
#pragma unroll
      for (int gate = 0; gate < 4; ++gate) {
        float b = bias1[gate * 128 + g8 * 16 + lr];
#pragma unroll
        for (int e = 0; e < 8; ++e) acc[gate][e] = b;
        int nt = gate * 8 + g8;
#pragma unroll
        for (int kt = 0; kt < 4; ++kt) {
          v16h bf = *(const v16h*)&whh_lds[(((nt * 4) + kt) * 32 + lane) * 16];
          acc[gate] = __builtin_amdgcn_wmma_f32_16x16x32_f16(
              false, ah[kt].v, false, bf, (short)0, acc[gate], false, false);
        }
#pragma unroll
        for (int kt = 0; kt < 2; ++kt) {
          v16h bf = *(const v16h*)&wih_lds[(((nt * 2) + kt) * 32 + lane) * 16];
          acc[gate] = __builtin_amdgcn_wmma_f32_16x16x32_f16(
              false, ax[kt].v, false, bf, (short)0, acc[gate], false, false);
        }
      }
      // LSTM gates on C-fragment layout: VGPR e -> row M=e+8*hi, col N=g8*16+lr
#pragma unroll
      for (int e = 0; e < 8; ++e) {
        float zi = acc[0][e], zf = acc[1][e], zg = acc[2][e], zo = acc[3][e];
        float c = sigm(zf) * cf[g8][e] + sigm(zi) * tanhf(zg);
        cf[g8][e] = c;
        float h = sigm(zo) * tanhf(c);
        int row = e + 8 * hi, col = g8 * 16 + lr;
        h_lds[row * 128 + col] = (_Float16)h;
        if (t == CST - 1) h1out[(seq0 + row) * 128 + col] = h;
      }
    }
  }
}

// ---------------------------------------------------------------------------
// last_pre = relu(h1 @ Wc^T + bc)  — one block per row, 128 threads
// ---------------------------------------------------------------------------
__global__ void k_rowrelu_wc(const float* __restrict__ X, const float* __restrict__ Wc,
                             const float* __restrict__ bc, float* __restrict__ Y) {
  __shared__ float xs[128];
  size_t m = blockIdx.x; int j = threadIdx.x;
  xs[j] = X[m * 128 + j];
  __syncthreads();
  float s = bc[j];
#pragma unroll 8
  for (int k = 0; k < 128; ++k) s += xs[k] * Wc[j * 128 + k];
  Y[m * 128 + j] = fmaxf(s, 0.f);
}

// per-column mean/var over M rows (ld = 128); one block per column
__global__ void k_colstats(const float* __restrict__ X, int M,
                           float* __restrict__ mean, float* __restrict__ var) {
  __shared__ float ssum[256], ssq[256];
  int c = blockIdx.x, t = threadIdx.x;
  float s = 0.f, q = 0.f;
  for (int r = t; r < M; r += 256) { float v = X[(size_t)r * 128 + c]; s += v; q += v * v; }
  ssum[t] = s; ssq[t] = q; __syncthreads();
  for (int o = 128; o > 0; o >>= 1) {
    if (t < o) { ssum[t] += ssum[t + o]; ssq[t] += ssq[t + o]; }
    __syncthreads();
  }
  if (t == 0) { float mu = ssum[0] / M; mean[c] = mu; var[c] = ssq[0] / M - mu * mu; }
}

// feats f16 [L*N][192] = [operators(4) | extra(56) | BN1(last_pre)(128) | pad(4)]
__global__ void k_feats(const float* __restrict__ op, const float* __restrict__ ex,
                        const float* __restrict__ lastpre,
                        const float* __restrict__ g1, const float* __restrict__ b1,
                        const float* __restrict__ mean, const float* __restrict__ var,
                        _Float16* __restrict__ feats) {
  size_t total = (size_t)SEQ * FEATW;
  for (size_t i = (size_t)blockIdx.x * blockDim.x + threadIdx.x; i < total;
       i += (size_t)gridDim.x * blockDim.x) {
    size_t row = i / FEATW; int c = (int)(i % FEATW);
    float v = 0.f;
    if (c < 4)        v = op[row * 4 + c];
    else if (c < 60)  v = ex[row * 56 + (c - 4)];
    else if (c < 188) {
      int k = c - 60; float x = lastpre[row * 128 + k];
      v = (x - mean[k]) * rsqrtf(var[k] + 1e-5f) * g1[k] + b1[k];
    }
    feats[i] = (_Float16)v;
  }
}

// ---------------------------------------------------------------------------
// Phase 2: one tree level. 32 blocks x 8 waves; wave owns 16 rows of N=4096.
// z = [x(192)|avg_h(128)] @ W2cat^T (K=320, 10 f16 WMMA k-tiles), gates, h/c out.
// ---------------------------------------------------------------------------
__global__ void __launch_bounds__(256) k_level(
    const _Float16* __restrict__ feats,   // [4096][192] level slice
    const int* __restrict__ mapping,      // [4096][2] (unused when bottom)
    const float* __restrict__ h_prev, const float* __restrict__ c_prev,
    const _Float16* __restrict__ w2pack,  // 32nt x 10kt frag layout (L2-resident)
    const float* __restrict__ bias2,
    float* __restrict__ h_out, float* __restrict__ c_out, int bottom)
{
  extern __shared__ char smem[];
  int tid = threadIdx.x, wave = tid >> 5, lane = tid & 31;
  int hi = lane >> 4, lr = lane & 15;
  _Float16* xcat = (_Float16*)smem + wave * (16 * KCAT);
  float* avgc = (float*)(smem + 8 * 16 * KCAT * 2) + wave * (16 * 128);
  int row0 = (blockIdx.x * 8 + wave) * 16;

  for (int i = lane; i < 16 * FEATW; i += 32) {
    int r = i / FEATW, c = i % FEATW;
    xcat[r * KCAT + c] = feats[(size_t)(row0 + r) * FEATW + c];
  }
  if (bottom) {
    for (int i = lane; i < 16 * 128; i += 32) {
      int r = i >> 7, c = i & 127;
      xcat[r * KCAT + 192 + c] = (_Float16)0.f;
      avgc[r * 128 + c] = 0.f;
    }
  } else {
    for (int i = lane; i < 16 * 128; i += 32) {
      int r = i >> 7, c = i & 127;
      int m0 = mapping[(size_t)(row0 + r) * 2 + 0];
      int m1 = mapping[(size_t)(row0 + r) * 2 + 1];
      float hv = 0.f, cv = 0.f;
      if (m0 > 0) { hv += h_prev[(size_t)(m0 - 1) * 128 + c]; cv += c_prev[(size_t)(m0 - 1) * 128 + c]; }
      if (m1 > 0) { hv += h_prev[(size_t)(m1 - 1) * 128 + c]; cv += c_prev[(size_t)(m1 - 1) * 128 + c]; }
      xcat[r * KCAT + 192 + c] = (_Float16)(0.5f * hv);
      avgc[r * 128 + c] = 0.5f * cv;
    }
  }
  // A-fragments for K=320 (10 k-tiles) — intra-wave LDS dependency only
  AF a[10];
#pragma unroll
  for (int kt = 0; kt < 10; ++kt)
#pragma unroll
    for (int p = 0; p < 8; ++p) {
      int k = kt * 32 + 2 * p + 8 * hi + ((p >= 4) ? 8 : 0);
      a[kt].u[p] = *(const unsigned*)&xcat[lr * KCAT + k];
    }

#pragma unroll
  for (int g8 = 0; g8 < 8; ++g8) {
    v8f acc[4];
#pragma unroll
    for (int gate = 0; gate < 4; ++gate) {
      float b = bias2[gate * 128 + g8 * 16 + lr];
#pragma unroll
      for (int e = 0; e < 8; ++e) acc[gate][e] = b;
      int nt = gate * 8 + g8;
#pragma unroll
      for (int kt = 0; kt < 10; ++kt) {
        const _Float16* bp = &w2pack[(size_t)(((nt * 10) + kt) * 32 + lane) * 16];
        if (kt < 9) __builtin_prefetch(bp + 32 * 16, 0, 0);   // global_prefetch_b8
        v16h bf = *(const v16h*)bp;
        acc[gate] = __builtin_amdgcn_wmma_f32_16x16x32_f16(
            false, a[kt].v, false, bf, (short)0, acc[gate], false, false);
      }
    }
#pragma unroll
    for (int e = 0; e < 8; ++e) {
      float zi = acc[0][e], zf = acc[1][e], zg = acc[2][e], zo = acc[3][e];
      int row = e + 8 * hi, col = g8 * 16 + lr;
      float c = sigm(zf) * avgc[row * 128 + col] + sigm(zi) * tanhf(zg);
      float h = sigm(zo) * tanhf(c);
      h_out[(size_t)(row0 + row) * 128 + col] = h;
      c_out[(size_t)(row0 + row) * 128 + col] = c;
    }
  }
}

// Generic head stage: optional input BN, y = act(W@x + b); block per row
__global__ void k_mlp(const float* __restrict__ X, const float* __restrict__ W,
                      const float* __restrict__ b,
                      const float* __restrict__ g, const float* __restrict__ beta,
                      const float* __restrict__ mean, const float* __restrict__ var,
                      float* __restrict__ Y, int Nout, int act) {
  __shared__ float xs[128];
  size_t m = blockIdx.x; int j = threadIdx.x;
  float x = X[m * 128 + j];
  if (g) x = (x - mean[j]) * rsqrtf(var[j] + 1e-5f) * g[j] + beta[j];
  xs[j] = x;
  __syncthreads();
  if (j < Nout) {
    float s = b[j];
#pragma unroll 8
    for (int k = 0; k < 128; ++k) s += xs[k] * W[j * 128 + k];
    s = (act == 0) ? fmaxf(s, 0.f) : 1.f / (1.f + __expf(-s));
    Y[m * Nout + j] = s;
  }
}

// ---------------------------------------------------------------------------
extern "C" void kernel_launch(void* const* d_in, const int* in_sizes, int n_in,
                              void* d_out, int out_size, void* d_ws, size_t ws_size,
                              hipStream_t stream) {
  const float* operators = (const float*)d_in[0];
  const float* extra     = (const float*)d_in[1];
  const float* cond      = (const float*)d_in[2];
  const int*   mapping   = (const int*)d_in[3];
  const float* Wih1 = (const float*)d_in[4];  const float* Whh1 = (const float*)d_in[5];
  const float* bih1 = (const float*)d_in[6];  const float* bhh1 = (const float*)d_in[7];
  const float* Wc   = (const float*)d_in[8];  const float* bc   = (const float*)d_in[9];
  const float* g1   = (const float*)d_in[10]; const float* be1  = (const float*)d_in[11];
  const float* Wih2 = (const float*)d_in[12]; const float* Whh2 = (const float*)d_in[13];
  const float* bih2 = (const float*)d_in[14]; const float* bhh2 = (const float*)d_in[15];
  const float* g2   = (const float*)d_in[16]; const float* be2  = (const float*)d_in[17];
  const float* W2   = (const float*)d_in[18]; const float* b2   = (const float*)d_in[19];
  const float* g3   = (const float*)d_in[20]; const float* be3  = (const float*)d_in[21];
  const float* W3   = (const float*)d_in[22]; const float* b3   = (const float*)d_in[23];
  const float* Wo   = (const float*)d_in[24]; const float* bo   = (const float*)d_in[25];

  char* base = (char*)d_ws;
  size_t off = 0;
  auto alloc = [&](size_t bytes) { size_t r = off; off = (off + bytes + 255) & ~(size_t)255; return r; };
  size_t o_wih1p = alloc((size_t)32 * 2 * 32 * 16 * 2);
  size_t o_whh1p = alloc((size_t)32 * 4 * 32 * 16 * 2);
  size_t o_w2p   = alloc((size_t)32 * 10 * 32 * 16 * 2);
  size_t o_bia1  = alloc(512 * 4);
  size_t o_bia2  = alloc(512 * 4);
  size_t o_h1    = alloc((size_t)SEQ * 128 * 4);
  size_t o_last  = alloc((size_t)SEQ * 128 * 4);
  size_t o_feats = alloc((size_t)SEQ * FEATW * 2);
  size_t o_hA = alloc((size_t)NN * 128 * 4), o_cA = alloc((size_t)NN * 128 * 4);
  size_t o_hB = alloc((size_t)NN * 128 * 4), o_cB = alloc((size_t)NN * 128 * 4);
  size_t o_t1 = alloc((size_t)NN * 128 * 4), o_t2 = alloc((size_t)NN * 128 * 4);
  size_t o_t3 = alloc((size_t)NN * 128 * 4);
  size_t o_m1 = alloc(128 * 4), o_v1 = alloc(128 * 4);
  size_t o_m2 = alloc(128 * 4), o_v2 = alloc(128 * 4);
  size_t o_m3 = alloc(128 * 4), o_v3 = alloc(128 * 4);
  (void)ws_size; (void)n_in; (void)in_sizes; (void)out_size;

  _Float16* wih1p = (_Float16*)(base + o_wih1p);
  _Float16* whh1p = (_Float16*)(base + o_whh1p);
  _Float16* w2p   = (_Float16*)(base + o_w2p);
  float* bia1 = (float*)(base + o_bia1);
  float* bia2 = (float*)(base + o_bia2);
  float* h1   = (float*)(base + o_h1);
  float* lastp = (float*)(base + o_last);
  _Float16* featsp = (_Float16*)(base + o_feats);
  float* hA = (float*)(base + o_hA); float* cA = (float*)(base + o_cA);
  float* hB = (float*)(base + o_hB); float* cB = (float*)(base + o_cB);
  float* t1 = (float*)(base + o_t1); float* t2 = (float*)(base + o_t2);
  float* t3 = (float*)(base + o_t3);
  float* m1 = (float*)(base + o_m1); float* v1 = (float*)(base + o_v1);
  float* m2 = (float*)(base + o_m2); float* v2 = (float*)(base + o_v2);
  float* m3 = (float*)(base + o_m3); float* v3 = (float*)(base + o_v3);

  // weight packing
  k_pack_frag<<<64, 256, 0, stream>>>(Wih1, 64, 32, 2, wih1p);
  k_pack_frag<<<64, 256, 0, stream>>>(Whh1, 128, 32, 4, whh1p);
  k_pack_w2<<<160, 256, 0, stream>>>(Wih2, Whh2, w2p);
  k_addvec<<<2, 256, 0, stream>>>(bih1, bhh1, bia1, 512);
  k_addvec<<<2, 256, 0, stream>>>(bih2, bhh2, bia2, 512);

  // Phase 1: fused 10-step LSTM, 768 blocks x 8 waves x 16 seq = 98304 sequences
  k_lstm1<<<SEQ / 128, 256, P1_SMEM, stream>>>(cond, wih1p, whh1p, bia1, h1);

  // last = BN1(relu(h1 @ Wc^T + bc)); feats assembly
  k_rowrelu_wc<<<SEQ, 128, 0, stream>>>(h1, Wc, bc, lastp);
  k_colstats<<<128, 256, 0, stream>>>(lastp, SEQ, m1, v1);
  k_feats<<<4096, 256, 0, stream>>>(operators, extra, lastp, g1, be1, m1, v1, featsp);

  // Phase 2: bottom-up tree recurrence, 24 sequential levels
  float* hprev = nullptr; float* cprev = nullptr;
  for (int lev = LVL - 1; lev >= 0; --lev) {
    int parity = (LVL - 1 - lev) & 1;
    float* ho = parity ? hB : hA;
    float* co = parity ? cB : cA;
    k_level<<<32, 256, P2_SMEM, stream>>>(
        featsp + (size_t)lev * NN * FEATW, mapping + (size_t)lev * NN * 2,
        hprev, cprev, w2p, bia2, ho, co, (lev == LVL - 1) ? 1 : 0);
    hprev = ho; cprev = co;
  }

  // Head: BN2 -> relu(W2) -> BN3 -> relu(W3) -> relu(W3) -> sigmoid(Wo)
  k_colstats<<<128, 256, 0, stream>>>(hprev, NN, m2, v2);
  k_mlp<<<NN, 128, 0, stream>>>(hprev, W2, b2, g2, be2, m2, v2, t1, 128, 0);
  k_colstats<<<128, 256, 0, stream>>>(t1, NN, m3, v3);
  k_mlp<<<NN, 128, 0, stream>>>(t1, W3, b3, g3, be3, m3, v3, t2, 128, 0);
  k_mlp<<<NN, 128, 0, stream>>>(t2, W3, b3, nullptr, nullptr, nullptr, nullptr, t3, 128, 0);
  k_mlp<<<NN, 128, 0, stream>>>(t3, Wo, bo, nullptr, nullptr, nullptr, nullptr,
                                (float*)d_out, 1, 1);
}
